// BaseReservoirExpert_4123168604304
// MI455X (gfx1250) — compile-verified, matrix operand order checked by reading, not run on hardware
//
#include <hip/hip_runtime.h>
#include <hip/hip_bf16.h>

// ---------------------------------------------------------------------------
// Leaky ESN on MI455X (gfx1250, wave32, WMMA).
//
//   h_{t+1} = (1-a) h_t + a * tanh( h_t @ W_res^T + x_t @ W_in^T )
//
// Fused input projection: per-step GEMM with concatenated K = 2048 + 512,
// bf16 WMMA with f32 accumulate. W_res (8 MB bf16) stays resident in the
// 192 MB L2 across all 512 sequential steps, so each step is latency-bound,
// not bandwidth-bound: maximize WGP spread (128 blocks x 2 waves = 1 wave
// per SIMD) and pipeline loads deeply (unroll 4 -> up to 16 outstanding
// global_load_b128 per wave).
// ---------------------------------------------------------------------------

#define B_SZ 32
#define T_SZ 512
#define D_SZ 512
#define R_SZ 2048
#define LEAK 0.25f

typedef __attribute__((ext_vector_type(16))) __bf16 v16bf;
typedef __attribute__((ext_vector_type(8)))  float  v8f;

static __device__ __forceinline__ unsigned short f32_to_bf16_rne(float f) {
    unsigned int u = __float_as_uint(f);
    u += 0x7FFFu + ((u >> 16) & 1u);   // round-to-nearest-even
    return (unsigned short)(u >> 16);
}

// ---- one-time converts ----------------------------------------------------

__global__ void cvt_f32_bf16(const float* __restrict__ in,
                             unsigned short* __restrict__ out, int n) {
    int i = blockIdx.x * blockDim.x + threadIdx.x;
    if (i < n) out[i] = f32_to_bf16_rne(in[i]);
}

// x[b][t][d] (f32) -> xt[t][b][d] (bf16): contiguous d per (t,b) row for
// WMMA A-operand loads of the fused input-projection K phase.
__global__ void cvt_x_tbd(const float* __restrict__ x,
                          unsigned short* __restrict__ xt) {
    int i = blockIdx.x * blockDim.x + threadIdx.x;   // flat over [b][t][d]
    if (i >= B_SZ * T_SZ * D_SZ) return;
    int d = i & (D_SZ - 1);
    int t = (i >> 9) & (T_SZ - 1);
    int b = i >> 18;
    xt[((size_t)t * B_SZ + b) * D_SZ + d] = f32_to_bf16_rne(x[i]);
}

__global__ void zero_h(float* __restrict__ hf,
                       unsigned short* __restrict__ hb, int n) {
    int i = blockIdx.x * blockDim.x + threadIdx.x;
    if (i < n) { hf[i] = 0.0f; hb[i] = 0; }
}

// ---- one ESN timestep -----------------------------------------------------
// Grid: 128 blocks x 64 threads (2 waves). Wave w owns the 16x16 output tile
// (m0 = w*16, n0 = blockIdx.x*16): 64 WMMAs over h@W_res^T (K=2048), then
// 16 WMMAs over x_t@W_in^T (K=512), then leak+tanh epilogue. One wave per
// SIMD keeps the matrix pipe uncontended; the accumulate chain issues
// back-to-back while unrolled b128 loads hide L2 latency.
//
// 16-bit operand striping per ISA 7.12.2: lanes 0-15 hold row (lane), K
// octets {0..7, 16..23}; lanes 16-31 hold row (lane-16), K octets
// {8..15, 24..31}. Both A rows (h / x_t) and B columns (W_res / W_in rows,
// since B[k][n] = W[n][k]) are contiguous in K in our layouts, so each tile
// operand is two 16-byte loads per lane.

__global__ __launch_bounds__(64) void esn_step(
    const unsigned short* __restrict__ h_in_bf,   // [32][2048]
    const float*          __restrict__ h_in_f32,  // [32][2048]
    const unsigned short* __restrict__ x_t_bf,    // [32][512]  (slice at t)
    const unsigned short* __restrict__ w_res_bf,  // [2048][2048]
    const unsigned short* __restrict__ w_in_bf,   // [2048][512]
    float*                __restrict__ h_out_f32,
    unsigned short*       __restrict__ h_out_bf)
{
    const int lane = threadIdx.x & 31;
    const int wave = threadIdx.x >> 5;          // 0..1
    const int m0   = wave * 16;                 // batch-row tile base
    const int n0   = blockIdx.x * 16;           // reservoir-column tile base

    const int rlo  = lane & 15;
    const int koff = (lane < 16) ? 0 : 8;

    const unsigned short* aBase = h_in_bf  + (size_t)(m0 + rlo) * R_SZ + koff;
    const unsigned short* bBase = w_res_bf + (size_t)(n0 + rlo) * R_SZ + koff;

    v8f acc = {};

    // Phase 1: h @ W_res^T  (K = 2048)
    #pragma unroll 4
    for (int k0 = 0; k0 < R_SZ; k0 += 32) {
        union { uint4 u[2]; v16bf v; } a, b;
        a.u[0] = *(const uint4*)(aBase + k0);
        a.u[1] = *(const uint4*)(aBase + k0 + 16);
        b.u[0] = *(const uint4*)(bBase + k0);
        b.u[1] = *(const uint4*)(bBase + k0 + 16);
        __builtin_prefetch(bBase + k0 + 256, 0, 3);   // near-scope prefetch
        acc = __builtin_amdgcn_wmma_f32_16x16x32_bf16(
            false, a.v, false, b.v, (short)0, acc, false, false);
    }

    // Phase 2: x_t @ W_in^T  (K = 512), fused input projection
    const unsigned short* aBase2 = x_t_bf  + (size_t)(m0 + rlo) * D_SZ + koff;
    const unsigned short* bBase2 = w_in_bf + (size_t)(n0 + rlo) * D_SZ + koff;
    #pragma unroll 4
    for (int k0 = 0; k0 < D_SZ; k0 += 32) {
        union { uint4 u[2]; v16bf v; } a, b;
        a.u[0] = *(const uint4*)(aBase2 + k0);
        a.u[1] = *(const uint4*)(aBase2 + k0 + 16);
        b.u[0] = *(const uint4*)(bBase2 + k0);
        b.u[1] = *(const uint4*)(bBase2 + k0 + 16);
        acc = __builtin_amdgcn_wmma_f32_16x16x32_bf16(
            false, a.v, false, b.v, (short)0, acc, false, false);
    }

    // Epilogue. D layout (32-bit C/D 16x16): VGPR j, lanes 0-15 -> (M=j,
    // N=lane); lanes 16-31 -> (M=j+8, N=lane-16).
    const int n   = n0 + rlo;
    const int mhi = (lane >> 4) * 8;
    #pragma unroll
    for (int j = 0; j < 8; ++j) {
        int m = m0 + j + mhi;
        size_t idx = (size_t)m * R_SZ + n;
        float hn = (1.0f - LEAK) * h_in_f32[idx] + LEAK * tanhf(acc[j]);
        h_out_f32[idx] = hn;
        h_out_bf[idx]  = f32_to_bf16_rne(hn);
    }
}

// ---------------------------------------------------------------------------

extern "C" void kernel_launch(void* const* d_in, const int* in_sizes, int n_in,
                              void* d_out, int out_size, void* d_ws, size_t ws_size,
                              hipStream_t stream) {
    const float* x     = (const float*)d_in[0];   // [32][512][512]
    const float* W_in  = (const float*)d_in[1];   // [2048][512]
    const float* W_res = (const float*)d_in[2];   // [2048][2048]

    char* ws = (char*)d_ws;
    size_t off = 0;
    unsigned short* w_res_bf = (unsigned short*)(ws + off); off += (size_t)R_SZ * R_SZ * 2;       // 8 MB
    unsigned short* w_in_bf  = (unsigned short*)(ws + off); off += (size_t)R_SZ * D_SZ * 2;       // 2 MB
    unsigned short* x_bf     = (unsigned short*)(ws + off); off += (size_t)B_SZ * T_SZ * D_SZ * 2;// 16 MB
    float*          h_f32[2];
    unsigned short* h_bf[2];
    h_f32[0] = (float*)(ws + off);          off += (size_t)B_SZ * R_SZ * 4;
    h_f32[1] = (float*)(ws + off);          off += (size_t)B_SZ * R_SZ * 4;
    h_bf[0]  = (unsigned short*)(ws + off); off += (size_t)B_SZ * R_SZ * 2;
    h_bf[1]  = (unsigned short*)(ws + off); off += (size_t)B_SZ * R_SZ * 2;

    // One-time precision/layout converts (cheap; streams ~50 MB once).
    {
        int n = R_SZ * R_SZ;
        cvt_f32_bf16<<<(n + 255) / 256, 256, 0, stream>>>(W_res, w_res_bf, n);
        n = R_SZ * D_SZ;
        cvt_f32_bf16<<<(n + 255) / 256, 256, 0, stream>>>(W_in, w_in_bf, n);
        n = B_SZ * T_SZ * D_SZ;
        cvt_x_tbd<<<(n + 255) / 256, 256, 0, stream>>>(x, x_bf);
        n = B_SZ * R_SZ;
        zero_h<<<(n + 255) / 256, 256, 0, stream>>>(h_f32[0], h_bf[0], n);
    }

    // Sequential scan: 512 small fused GEMM steps, ping-pong h buffers.
    int cur = 0;
    for (int t = 0; t < T_SZ; ++t) {
        esn_step<<<R_SZ / 16, 64, 0, stream>>>(
            h_bf[cur], h_f32[cur],
            x_bf + (size_t)t * B_SZ * D_SZ,
            w_res_bf, w_in_bf,
            h_f32[1 - cur], h_bf[1 - cur]);
        cur ^= 1;
    }

    hipMemcpyAsync(d_out, h_f32[cur], (size_t)B_SZ * R_SZ * sizeof(float),
                   hipMemcpyDeviceToDevice, stream);
}
